// VisionTransformerEnsemblePostCrossAttention_17111149707391
// MI455X (gfx1250) — matrix operand order repaired and checked by reading, not compiled, not run on hardware
//
#include <hip/hip_runtime.h>
#include <math.h>

// ---------------------------------------------------------------------------
// Types for CDNA5 WMMA (wave32, 16x16x32 bf16 -> f32)
// ---------------------------------------------------------------------------
typedef __attribute__((ext_vector_type(16))) __bf16 v16bf;
typedef __attribute__((ext_vector_type(8)))  float  v8f;

union Frag {
    v16bf v;
    unsigned int u[8];
};

#define BM 128
#define BN 128
#define BK 32
#define LP 40   // LDS row pitch (bf16 elems): 80B -> 16B aligned, conflict-free

// f32 -> bf16 round-to-nearest-even
__device__ __forceinline__ unsigned short f2bf(float f) {
    unsigned int x = __float_as_uint(f);
    x += 0x7fffu + ((x >> 16) & 1u);
    return (unsigned short)(x >> 16);
}
__device__ __forceinline__ unsigned int pack2(float lo, float hi) {
    return (unsigned int)f2bf(lo) | ((unsigned int)f2bf(hi) << 16);
}

// ---------------------------------------------------------------------------
// Generic batched GEMM:  C[z] = op(A[z] @ B[z]) with flags:
//   bit0 = add bias[n], bit1 = accumulate into C, bit2 = exact GELU,
//   bit3 = B is [N,K] (transposed access)
// z = blockIdx.z; b = z / H; h = z % H; per-(b,h) pointer strides.
// 256 threads = 8 waves; 128x128 tile; each wave owns a 32x64 quadrant
// (2 A-fragments x 4 B-fragments -> 8 WMMA per 32-K step, B frags reused).
// Full interior tiles take an unpredicated float4-load fast path.
// ---------------------------------------------------------------------------
__global__ void __launch_bounds__(256)
gemm_wmma_kernel(const float* __restrict__ A, const float* __restrict__ Bm,
                 const float* __restrict__ bias, float* __restrict__ C,
                 int M, int N, int K, int lda, int ldb, int ldc, int H,
                 long long sAb, long long sAh, long long sBb, long long sBh,
                 long long sCb, long long sCh, int flags)
{
    __shared__ unsigned short As[BM * LP];   // As[m][k]
    __shared__ unsigned short Bs[BN * LP];   // Bs[n][k]

    const int z  = blockIdx.z;
    const int bb = z / H, hh = z % H;
    const float* Ab = A  + bb * sAb + hh * sAh;
    const float* Bb = Bm + bb * sBb + hh * sBh;
    float*       Cb = C  + bb * sCb + hh * sCh;

    const int m0 = blockIdx.y * BM, n0 = blockIdx.x * BN;
    const int tid  = threadIdx.x;
    const int wave = tid >> 5, lane = tid & 31;
    const int wr = wave >> 1, wc = wave & 1;     // 4x2 wave grid of 32x64 tiles

    const bool fullMN = (m0 + BM <= M) && (n0 + BN <= N);

    v8f zero = {0.f,0.f,0.f,0.f,0.f,0.f,0.f,0.f};
    v8f acc[2][4] = {{zero,zero,zero,zero},{zero,zero,zero,zero}};

    const int lan = lane & 15;
    const int hlf = lane >> 4;
    const int akb = hlf * 8;     // A frag K base (lanes 16-31: +8)
    const int bkb = hlf * 8;     // B frag K base (lanes 16-31: +8)

    for (int k0 = 0; k0 < K; k0 += BK) {
        __syncthreads();
        if (fullMN && (k0 + BK <= K)) {
            // ---- fast path: unpredicated vector loads ----
            #pragma unroll
            for (int i = tid; i < (BM * BK) / 4; i += 256) {
                int row = i >> 3, kk = (i & 7) << 2;
                float4 vv = *(const float4*)(Ab + (long long)(m0 + row) * lda + (k0 + kk));
                *(uint2*)&As[row * LP + kk] = make_uint2(pack2(vv.x, vv.y), pack2(vv.z, vv.w));
            }
            if (flags & 8) {
                #pragma unroll
                for (int i = tid; i < (BN * BK) / 4; i += 256) {
                    int row = i >> 3, kk = (i & 7) << 2;
                    float4 vv = *(const float4*)(Bb + (long long)(n0 + row) * ldb + (k0 + kk));
                    *(uint2*)&Bs[row * LP + kk] = make_uint2(pack2(vv.x, vv.y), pack2(vv.z, vv.w));
                }
            } else {
                #pragma unroll
                for (int i = tid; i < (BN * BK) / 4; i += 256) {
                    int k = i >> 5, ng = (i & 31) << 2;
                    float4 vv = *(const float4*)(Bb + (long long)(k0 + k) * ldb + (n0 + ng));
                    Bs[(ng + 0) * LP + k] = f2bf(vv.x);
                    Bs[(ng + 1) * LP + k] = f2bf(vv.y);
                    Bs[(ng + 2) * LP + k] = f2bf(vv.z);
                    Bs[(ng + 3) * LP + k] = f2bf(vv.w);
                }
            }
        } else {
            // ---- edge path: per-element, zero-padded ----
            for (int i = tid; i < BM * BK; i += 256) {
                int row = i >> 5, k = i & 31;
                int gm = m0 + row, gk = k0 + k;
                float val = (gm < M && gk < K) ? Ab[(long long)gm * lda + gk] : 0.f;
                As[row * LP + k] = f2bf(val);
            }
            for (int i = tid; i < BN * BK; i += 256) {
                int n = i >> 5, k = i & 31;
                int gn = n0 + n, gk = k0 + k;
                float val = 0.f;
                if (gn < N && gk < K)
                    val = (flags & 8) ? Bb[(long long)gn * ldb + gk]
                                      : Bb[(long long)gk * ldb + gn];
                Bs[n * LP + k] = f2bf(val);
            }
        }
        __syncthreads();

        // A fragments: VGPR j holds K = (j&3)*2 + (j>>2)*16 + akb (, +1)
        Frag af[2];
        #pragma unroll
        for (int s = 0; s < 2; ++s) {
            int am = wr * 32 + s * 16 + lan;
            #pragma unroll
            for (int j = 0; j < 8; ++j) {
                int k = ((j & 3) << 1) + ((j >> 2) << 4) + akb;
                af[s].u[j] = *(const unsigned int*)&As[am * LP + k];
            }
        }
        // B fragments reused across both A rows
        #pragma unroll
        for (int nt = 0; nt < 4; ++nt) {
            Frag bf;
            int bn = wc * 64 + nt * 16 + lan;
            #pragma unroll
            for (int j = 0; j < 8; ++j)
                bf.u[j] = *(const unsigned int*)&Bs[bn * LP + bkb + ((j & 3) << 1) + ((j >> 2) << 4)];
            acc[0][nt] = __builtin_amdgcn_wmma_f32_16x16x32_bf16(
                false, af[0].v, false, bf.v, (short)0, acc[0][nt], false, false);
            acc[1][nt] = __builtin_amdgcn_wmma_f32_16x16x32_bf16(
                false, af[1].v, false, bf.v, (short)0, acc[1][nt], false, false);
        }
    }

    // Epilogue: C layout VGPR r -> M = r + 8*(lane>>4); N = lane&15
    if (fullMN) {
        #pragma unroll
        for (int s = 0; s < 2; ++s) {
            #pragma unroll
            for (int nt = 0; nt < 4; ++nt) {
                int gn = n0 + wc * 64 + nt * 16 + lan;
                #pragma unroll
                for (int r = 0; r < 8; ++r) {
                    int gm = m0 + wr * 32 + s * 16 + r + (hlf << 3);
                    float o = acc[s][nt][r];
                    if (flags & 1) o += bias[gn];
                    if (flags & 4) o = 0.5f * o * (1.0f + erff(o * 0.70710678118654752f));
                    long long ci = (long long)gm * ldc + gn;
                    if (flags & 2) Cb[ci] += o; else Cb[ci] = o;
                }
            }
        }
    } else {
        #pragma unroll
        for (int s = 0; s < 2; ++s) {
            #pragma unroll
            for (int nt = 0; nt < 4; ++nt) {
                int gn = n0 + wc * 64 + nt * 16 + lan;
                #pragma unroll
                for (int r = 0; r < 8; ++r) {
                    int gm = m0 + wr * 32 + s * 16 + r + (hlf << 3);
                    if (gm < M && gn < N) {
                        float o = acc[s][nt][r];
                        if (flags & 1) o += bias[gn];
                        if (flags & 4) o = 0.5f * o * (1.0f + erff(o * 0.70710678118654752f));
                        long long ci = (long long)gm * ldc + gn;
                        if (flags & 2) Cb[ci] += o; else Cb[ci] = o;
                    }
                }
            }
        }
    }
}

// ---------------------------------------------------------------------------
// LayerNorm: one wave32 per row (dim=768), eps=1e-6
// ---------------------------------------------------------------------------
__global__ void __launch_bounds__(128)
layernorm_kernel(const float* __restrict__ x, const float* __restrict__ s,
                 const float* __restrict__ b, float* __restrict__ y,
                 int rows, int dim)
{
    int row  = blockIdx.x * (blockDim.x >> 5) + (threadIdx.x >> 5);
    int lane = threadIdx.x & 31;
    if (row >= rows) return;
    const float* xr = x + (long long)row * dim;
    float sum = 0.f;
    for (int i = lane; i < dim; i += 32) sum += xr[i];
    #pragma unroll
    for (int o = 16; o; o >>= 1) sum += __shfl_xor(sum, o, 32);
    float mean = sum / dim;
    float vs = 0.f;
    for (int i = lane; i < dim; i += 32) { float d = xr[i] - mean; vs += d * d; }
    #pragma unroll
    for (int o = 16; o; o >>= 1) vs += __shfl_xor(vs, o, 32);
    float inv = rsqrtf(vs / dim + 1e-6f);
    float* yr = y + (long long)row * dim;
    for (int i = lane; i < dim; i += 32)
        yr[i] = (xr[i] - mean) * inv * s[i] + b[i];
}

// ---------------------------------------------------------------------------
// Row softmax with pre-scale: r = softmax(r * scale). One wave32 per row.
// ---------------------------------------------------------------------------
__global__ void __launch_bounds__(128)
softmax_kernel(float* __restrict__ S, int rows, int n, int ld, float scale)
{
    int row  = blockIdx.x * (blockDim.x >> 5) + (threadIdx.x >> 5);
    int lane = threadIdx.x & 31;
    if (row >= rows) return;
    float* r = S + (long long)row * ld;
    float mx = -1e30f;
    for (int i = lane; i < n; i += 32) mx = fmaxf(mx, r[i] * scale);
    #pragma unroll
    for (int o = 16; o; o >>= 1) mx = fmaxf(mx, __shfl_xor(mx, o, 32));
    float sum = 0.f;
    for (int i = lane; i < n; i += 32) {
        float e = __expf(r[i] * scale - mx);
        r[i] = e; sum += e;
    }
    #pragma unroll
    for (int o = 16; o; o >>= 1) sum += __shfl_xor(sum, o, 32);
    float inv = 1.f / sum;
    for (int i = lane; i < n; i += 32) r[i] *= inv;
}

// ---------------------------------------------------------------------------
// Patch extraction: x [B,3,224,224] -> p [B,196,768], 768 index = c*256+py*16+px
// ---------------------------------------------------------------------------
__global__ void patches_kernel(const float* __restrict__ x, float* __restrict__ p,
                               long long total)
{
    long long idx = (long long)blockIdx.x * blockDim.x + threadIdx.x;
    if (idx >= total) return;
    int d = (int)(idx % 768);
    long long r = idx / 768;
    int patch = (int)(r % 196);
    int b = (int)(r / 196);
    int c  = d >> 8;
    int py = (d >> 4) & 15;
    int px = d & 15;
    int gy = patch / 14, gx = patch % 14;
    int iy = gy * 16 + py, ix = gx * 16 + px;
    p[idx] = x[((long long)(b * 3 + c) * 224 + iy) * 224 + ix];
}

// ---------------------------------------------------------------------------
// CLS row + positional embedding: t[b,0,:] = cls + pos[0]; t[b,n,:] += pos[n]
// ---------------------------------------------------------------------------
__global__ void cls_pos_kernel(float* __restrict__ t, const float* __restrict__ cls,
                               const float* __restrict__ pos, long long total)
{
    long long idx = (long long)blockIdx.x * blockDim.x + threadIdx.x;
    if (idx >= total) return;
    int d = (int)(idx % 768);
    int n = (int)((idx / 768) % 197);
    if (n == 0) t[idx] = cls[d] + pos[d];
    else        t[idx] += pos[(long long)n * 768 + d];
}

// ---------------------------------------------------------------------------
// Head GEMV: out[b,n] = f[b,0,:] @ W[:,n] + bias[n]   (tiny: 8x1000x768)
// ---------------------------------------------------------------------------
__global__ void head_kernel(const float* __restrict__ f, const float* __restrict__ w,
                            const float* __restrict__ b, float* __restrict__ out,
                            int Bn)
{
    int idx = blockIdx.x * blockDim.x + threadIdx.x;
    if (idx >= Bn * 1000) return;
    int n = idx % 1000, bb = idx / 1000;
    const float* fr = f + (long long)bb * 197 * 768;
    float s = b[n];
    for (int d = 0; d < 768; ++d) s += fr[d] * w[(long long)d * 1000 + n];
    out[idx] = s;
}

// ---------------------------------------------------------------------------
// Host-side orchestration
// ---------------------------------------------------------------------------
static inline void launch_gemm(hipStream_t st, const float* A, const float* Bm,
                               const float* bias, float* C,
                               int M, int N, int K, int lda, int ldb, int ldc,
                               int Z, int H,
                               long long sAb, long long sAh, long long sBb,
                               long long sBh, long long sCb, long long sCh, int flags)
{
    dim3 grid((N + BN - 1) / BN, (M + BM - 1) / BM, Z);
    gemm_wmma_kernel<<<grid, 256, 0, st>>>(A, Bm, bias, C, M, N, K, lda, ldb, ldc,
                                           H, sAb, sAh, sBb, sBh, sCb, sCh, flags);
}

static inline void launch_ln(hipStream_t st, const float* x, const float* s,
                             const float* b, float* y, int rows)
{
    layernorm_kernel<<<(rows + 3) / 4, 128, 0, st>>>(x, s, b, y, rows, 768);
}

static inline void launch_softmax(hipStream_t st, float* S, int rows, int n,
                                  int ld, float scale)
{
    softmax_kernel<<<(rows + 3) / 4, 128, 0, st>>>(S, rows, n, ld, scale);
}

// Cross-attention (H=8, hd=96): out = att(q,q) + att(q,kv2[1:]) + att(q,kv3[1:])
static void launch_cross(hipStream_t st, const float* q, const float* s2,
                         const float* s3, float* outp, float* S)
{
    const int HC = 8, HD = 96, NT = 197;
    const float scale = 0.10206207261596575f;   // 96^-0.5
    const float* kv[3] = { q, s2 + 768, s3 + 768 };
    const int    nk[3] = { 197, 196, 196 };
    for (int i = 0; i < 3; ++i) {
        // S = Q @ KV^T  (batched over b=8, h=8)
        launch_gemm(st, q, kv[i], nullptr, S, NT, nk[i], HD, 768, 768, 256,
                    64, HC, 197LL * 768, 96, 197LL * 768, 96,
                    (long long)HC * 197 * 256, 197LL * 256, /*transB*/8);
        launch_softmax(st, S, 64 * NT, nk[i], 256, scale);
        // out (+)= P @ KV
        launch_gemm(st, S, kv[i], nullptr, outp, NT, HD, nk[i], 256, 768, 768,
                    64, HC, (long long)HC * 197 * 256, 197LL * 256,
                    197LL * 768, 96, 197LL * 768, 96, (i == 0) ? 0 : 2);
    }
}

extern "C" void kernel_launch(void* const* d_in, const int* in_sizes, int n_in,
                              void* d_out, int out_size, void* d_ws, size_t ws_size,
                              hipStream_t stream)
{
    (void)in_sizes; (void)n_in; (void)out_size; (void)ws_size;

    const int Bn = 8, Dm = 768, NT = 197, NP = 196, Lr = 12;
    const int M = Bn * NT;                 // 1576 token rows, flattened over batch

    auto in = [&](int i) { return (const float*)d_in[i]; };
    // Input flatten order (setup_inputs dict insertion order, depth-first):
    //   0:x1 1:x2 2:x3, then per ViT (base VB): patch_w, patch_b, cls, pos,
    //   ln1_s, ln1_b, qkv_w, qkv_b, proj_w, proj_b, ln2_s, ln2_b,
    //   fc1_w, fc1_b, fc2_w, fc2_b, head_w, head_b  (18 arrays),
    //   then wq1, wq2, wq3, wq2_1, wq2_2, wq2_3.
    const int VB[3] = { 3, 21, 39 };
    const int WQ = 57, WQ2 = 60;

    // Workspace layout (f32)
    float* w = (float*)d_ws;
    size_t off = 0;
    auto alloc = [&](size_t n) { float* p = w + off; off += (n + 63) & ~(size_t)63; return p; };
    float *t[3], *v[3], *f[3], *g[3];
    for (int i = 0; i < 3; ++i) t[i] = alloc((size_t)M * Dm);
    for (int i = 0; i < 3; ++i) v[i] = alloc((size_t)M * Dm);
    for (int i = 0; i < 3; ++i) f[i] = alloc((size_t)M * Dm);
    for (int i = 0; i < 3; ++i) g[i] = alloc((size_t)M * Dm);
    float* h    = alloc((size_t)M * Dm);
    float* attO = alloc((size_t)M * Dm);
    float* big  = alloc((size_t)M * 4 * Dm);          // patches / qkv / mlp scratch
    float* S    = alloc((size_t)96 * 197 * 256);      // attention scores (ld=256)

    // ========== Three ViT backbones ==========
    for (int vi = 0; vi < 3; ++vi) {
        const int vb = VB[vi];
        const float* x       = in(vi);
        const float* patch_w = in(vb + 0);
        const float* patch_b = in(vb + 1);
        const float* cls     = in(vb + 2);
        const float* pos     = in(vb + 3);

        // Patch embed: gather -> [B,196,768] then GEMM into t rows 1..196
        {
            long long total = (long long)Bn * NP * Dm;
            patches_kernel<<<(unsigned)((total + 255) / 256), 256, 0, stream>>>(x, big, total);
        }
        launch_gemm(stream, big, patch_w, patch_b, t[vi] + Dm,
                    NP, Dm, Dm, Dm, Dm, Dm,
                    Bn, 1, (long long)NP * Dm, 0, 0, 0, (long long)NT * Dm, 0, /*bias*/1);
        {
            long long total = (long long)Bn * NT * Dm;
            cls_pos_kernel<<<(unsigned)((total + 255) / 256), 256, 0, stream>>>(t[vi], cls, pos, total);
        }

        // 12 transformer blocks
        for (int l = 0; l < Lr; ++l) {
            const float* ln1s = in(vb + 4) + (size_t)l * Dm;
            const float* ln1b = in(vb + 5) + (size_t)l * Dm;
            const float* qkvW = in(vb + 6) + (size_t)l * Dm * 3 * Dm;
            const float* qkvB = in(vb + 7) + (size_t)l * 3 * Dm;
            const float* prjW = in(vb + 8) + (size_t)l * Dm * Dm;
            const float* prjB = in(vb + 9) + (size_t)l * Dm;
            const float* ln2s = in(vb + 10) + (size_t)l * Dm;
            const float* ln2b = in(vb + 11) + (size_t)l * Dm;
            const float* fc1W = in(vb + 12) + (size_t)l * Dm * 4 * Dm;
            const float* fc1B = in(vb + 13) + (size_t)l * 4 * Dm;
            const float* fc2W = in(vb + 14) + (size_t)l * 4 * Dm * Dm;
            const float* fc2B = in(vb + 15) + (size_t)l * Dm;

            // h = LN1(t); qkv = h @ Wqkv + b   -> big [1576, 2304]
            launch_ln(stream, t[vi], ln1s, ln1b, h, M);
            launch_gemm(stream, h, qkvW, qkvB, big, M, 3 * Dm, Dm, Dm, 3 * Dm, 3 * Dm,
                        1, 1, 0, 0, 0, 0, 0, 0, /*bias*/1);

            // Self-attention, batched over (b=8, h=12), hd=64
            // S = Q @ K^T
            launch_gemm(stream, big, big + Dm, nullptr, S, NT, NT, 64,
                        2304, 2304, 256, 96, 12,
                        197LL * 2304, 64, 197LL * 2304, 64,
                        12LL * 197 * 256, 197LL * 256, /*transB*/8);
            launch_softmax(stream, S, 96 * NT, NT, 256, 0.125f);   // 64^-0.5
            // attO = P @ V
            launch_gemm(stream, S, big + 2 * Dm, nullptr, attO, NT, 64, NT,
                        256, 2304, Dm, 96, 12,
                        12LL * 197 * 256, 197LL * 256, 197LL * 2304, 64,
                        197LL * Dm, 64, 0);
            // t += attO @ Wproj + b
            launch_gemm(stream, attO, prjW, prjB, t[vi], M, Dm, Dm, Dm, Dm, Dm,
                        1, 1, 0, 0, 0, 0, 0, 0, /*bias|accum*/3);

            // MLP: t += gelu(LN2(t) @ W1 + b1) @ W2 + b2
            launch_ln(stream, t[vi], ln2s, ln2b, h, M);
            launch_gemm(stream, h, fc1W, fc1B, big, M, 4 * Dm, Dm, Dm, 4 * Dm, 4 * Dm,
                        1, 1, 0, 0, 0, 0, 0, 0, /*bias|gelu*/5);
            launch_gemm(stream, big, fc2W, fc2B, t[vi], M, Dm, 4 * Dm, 4 * Dm, Dm, Dm,
                        1, 1, 0, 0, 0, 0, 0, 0, /*bias|accum*/3);
        }

        // v_i = t_i @ wq_i
        launch_gemm(stream, t[vi], in(WQ + vi), nullptr, v[vi], M, Dm, Dm, Dm, Dm, Dm,
                    1, 1, 0, 0, 0, 0, 0, 0, 0);
    }

    // ========== Cross-attention stage 1 ==========
    launch_cross(stream, v[0], v[1], v[2], f[0], S);
    launch_cross(stream, v[1], v[0], v[2], f[1], S);
    launch_cross(stream, v[2], v[0], v[1], f[2], S);

    // g_i = f_i @ wq2_i
    for (int i = 0; i < 3; ++i)
        launch_gemm(stream, f[i], in(WQ2 + i), nullptr, g[i], M, Dm, Dm, Dm, Dm, Dm,
                    1, 1, 0, 0, 0, 0, 0, 0, 0);

    // ========== Cross-attention stage 2 (reuse f as outputs) ==========
    launch_cross(stream, g[0], g[1], g[2], f[0], S);
    launch_cross(stream, g[1], g[0], g[2], f[1], S);
    launch_cross(stream, g[2], g[0], g[1], f[2], S);

    // ========== Classifier heads on CLS token ==========
    float* out = (float*)d_out;
    for (int i = 0; i < 3; ++i)
        head_kernel<<<(Bn * 1000 + 255) / 256, 256, 0, stream>>>(
            f[i], in(VB[i] + 16), in(VB[i] + 17), out + (size_t)i * Bn * 1000, Bn);
}
